// HybridMultiHeadAttention_13993003450669
// MI455X (gfx1250) — matrix operand order repaired and checked by reading, not compiled, not run on hardware
//
#include <hip/hip_runtime.h>
#include <hip/hip_bf16.h>
#include <stdint.h>

#define D_MODEL 1024
#define N_HEADS 16
#define N_PHYS  8
#define D_HEAD  64
#define BATCH   2
#define SEQ     2048
#define ROWS    (BATCH * SEQ)   // 4096

#define LDSA 72                 // padded LDS row stride (elems): 144B, 16B-aligned,
                                // 36-dword stride -> 16 lanes hit 16 distinct banks

typedef __attribute__((ext_vector_type(16))) __bf16 v16bf;
typedef __attribute__((ext_vector_type(8)))  float  v8f;
typedef __attribute__((ext_vector_type(4)))  int    i4v;
typedef __attribute__((address_space(1)))    i4v    i4v_g;  // global int4
typedef __attribute__((address_space(3)))    i4v    i4v_l;  // LDS int4

union AFrag { v16bf v; uint32_t u[8]; };

#if defined(__has_builtin)
#if __has_builtin(__builtin_amdgcn_global_load_async_to_lds_b128)
#define HAVE_ASYNC_LDS 1
#endif
#endif

__device__ __forceinline__ uint16_t f2bf(float f) {
  uint32_t u = __float_as_uint(f);
  u += 0x7FFFu + ((u >> 16) & 1u);   // round-to-nearest-even
  return (uint16_t)(u >> 16);
}

// ---------------------------------------------------------------- fp32 -> bf16
__global__ void cvt_f32_bf16(const float* __restrict__ in,
                             uint16_t* __restrict__ out, int n) {
  int i = blockIdx.x * blockDim.x + threadIdx.x;
  int stride = gridDim.x * blockDim.x;
  for (; i < n; i += stride) out[i] = f2bf(in[i]);
}

// ------------------------------------------- fp32 [R,C] -> bf16 transposed [C,R]
__global__ __launch_bounds__(256)
void transpose_cvt(const float* __restrict__ in, uint16_t* __restrict__ out,
                   int R, int C) {
  __shared__ float tile[32][33];
  int c0 = blockIdx.x * 32, r0 = blockIdx.y * 32;
  int tx = threadIdx.x & 31, ty = threadIdx.x >> 5;   // 8 rows of 32
#pragma unroll
  for (int yy = ty; yy < 32; yy += 8)
    tile[yy][tx] = in[(size_t)(r0 + yy) * C + c0 + tx];
  __syncthreads();
#pragma unroll
  for (int yy = ty; yy < 32; yy += 8)
    out[(size_t)(c0 + yy) * R + r0 + tx] = f2bf(tile[tx][yy]);
}

// ================================================================ shared GEMM
// 128x128 block tile, 8 waves (4x2), K-step 64, double-buffered LDS staged via
// GLOBAL_LOAD_ASYNC_TO_LDS (ASYNCcnt) when available. Both A and B(=W^T) tiles
// are straight 128-row x 64-elem copies into LDSA-padded rows.
__device__ __forceinline__ void stage_tile(const uint16_t* __restrict__ gbase,
                                           uint16_t* lbuf, int tid) {
  // 128 rows x 64 elems; 16B chunks: idx = q*256+tid, row=idx>>3, ch=idx&7
#pragma unroll
  for (int q = 0; q < 4; ++q) {
    int idx = q * 256 + tid;
    int row = idx >> 3, ch = idx & 7;
    const uint16_t* g = gbase + (size_t)row * D_MODEL + ch * 8;
    uint16_t* l = lbuf + row * LDSA + ch * 8;
#ifdef HAVE_ASYNC_LDS
    // AS1 ptr: flat==global (identity). AS3 ptr: flat LDS addr keeps the LDS
    // byte offset in bits [31:0] (ISA 10.2), so truncation yields the offset.
    __builtin_amdgcn_global_load_async_to_lds_b128(
        (i4v_g*)(uintptr_t)g,
        (i4v_l*)(uint32_t)(uintptr_t)l,
        0, 0);
#else
    *reinterpret_cast<uint4*>(l) = *reinterpret_cast<const uint4*>(g);
#endif
  }
}

__device__ __forceinline__ void gemm_mainloop(
    const uint16_t* __restrict__ X, const uint16_t* __restrict__ Wt,
    uint16_t* As, uint16_t* Bt,    // each [2][128*LDSA]
    int rowBase, int colBase, v8f acc[2][4]) {
  const int tid  = threadIdx.x;
  const int wave = tid >> 5, lane = tid & 31;
  const int lrow = lane & 15, khalf = lane >> 4;
  const int wm = wave & 3, wn = wave >> 2;

  const uint16_t* Ag = X  + (size_t)rowBase * D_MODEL;
  const uint16_t* Bg = Wt + (size_t)colBase * D_MODEL;

  stage_tile(Ag, As, tid);
  stage_tile(Bg, Bt, tid);

  for (int i = 0; i < D_MODEL / 64; ++i) {
    const int p = i & 1;
    if (i + 1 < D_MODEL / 64) {
      stage_tile(Ag + (i + 1) * 64, As + (p ^ 1) * (128 * LDSA), tid);
      stage_tile(Bg + (i + 1) * 64, Bt + (p ^ 1) * (128 * LDSA), tid);
#ifdef HAVE_ASYNC_LDS
      asm volatile("s_wait_asynccnt 0x8" ::: "memory");  // prev tile's 8 ops done
#endif
    } else {
#ifdef HAVE_ASYNC_LDS
      asm volatile("s_wait_asynccnt 0x0" ::: "memory");
#endif
    }
    __syncthreads();

    const uint16_t* Ab = As + p * (128 * LDSA);
    const uint16_t* Bb = Bt + p * (128 * LDSA);
#pragma unroll
    for (int ks2 = 0; ks2 < 64; ks2 += 32) {
      AFrag a[2], b[4];
#pragma unroll
      for (int mi = 0; mi < 2; ++mi) {
        int mrow = wm * 32 + mi * 16 + lrow;
#pragma unroll
        for (int i2 = 0; i2 < 8; ++i2) {
          int k = ks2 + ((i2 >> 2) << 4) + (khalf << 3) + ((i2 & 3) << 1);
          a[mi].u[i2] = *reinterpret_cast<const uint32_t*>(Ab + mrow * LDSA + k);
        }
      }
#pragma unroll
      for (int ni = 0; ni < 4; ++ni) {
        int ncol = wn * 64 + ni * 16 + lrow;
#pragma unroll
        for (int j = 0; j < 8; ++j) {
          int k = ks2 + ((j >> 2) << 4) + (khalf << 3) + ((j & 3) << 1);
          b[ni].u[j] = *reinterpret_cast<const uint32_t*>(Bb + ncol * LDSA + k);
        }
      }
#pragma unroll
      for (int mi = 0; mi < 2; ++mi)
#pragma unroll
        for (int ni = 0; ni < 4; ++ni)
          acc[mi][ni] = __builtin_amdgcn_wmma_f32_16x16x32_bf16(
              false, a[mi].v, false, b[ni].v, (short)0, acc[mi][ni], false, false);
    }
    __syncthreads();
  }
}

// ---------------------------------------------------------------- QKV GEMM
__global__ __launch_bounds__(256)
void qkv_gemm(const uint16_t* __restrict__ X, const uint16_t* __restrict__ Wt,
              uint16_t* __restrict__ Q, uint16_t* __restrict__ K,
              uint16_t* __restrict__ VT) {
  __shared__ __align__(16) uint16_t As[2][128 * LDSA];
  __shared__ __align__(16) uint16_t Bt[2][128 * LDSA];

  const int tid  = threadIdx.x;
  const int wave = tid >> 5, lane = tid & 31;
  const int lrow = lane & 15, khalf = lane >> 4;
  const int wm = wave & 3, wn = wave >> 2;
  const int rowBase = blockIdx.y * 128;
  const int colBase = blockIdx.x * 128;

  v8f acc[2][4] = {};
  gemm_mainloop(X, Wt, As[0], Bt[0], rowBase, colBase, acc);

#pragma unroll
  for (int mi = 0; mi < 2; ++mi) {
    int row0 = rowBase + wm * 32 + mi * 16 + khalf * 8;  // 8 consecutive rows
    int b_ = row0 >> 11;
    int n0 = row0 & 2047;
#pragma unroll
    for (int ni = 0; ni < 4; ++ni) {
      int colT = colBase + wn * 64 + ni * 16;
      int seg = colT >> 10;
      int d0  = colT & 1023;
      int h   = d0 >> 6;
      int e   = (d0 & 63) + lrow;
      if (seg == 0) {
        uint16_t* dst = Q + ((size_t)(b_ * N_HEADS + h) * SEQ + n0) * D_HEAD + e;
#pragma unroll
        for (int r = 0; r < 8; ++r)
          dst[(size_t)r * D_HEAD] = f2bf(acc[mi][ni][r] * 0.125f);
      } else if (seg == 1) {
        uint16_t* dst = K + ((size_t)(b_ * N_HEADS + h) * SEQ + n0) * D_HEAD + e;
#pragma unroll
        for (int r = 0; r < 8; ++r)
          dst[(size_t)r * D_HEAD] = f2bf(acc[mi][ni][r]);
      } else {  // v transposed [b,h,d,n] -> contiguous in r
        uint16_t* dst = VT + ((size_t)(b_ * N_HEADS + h) * D_HEAD + e) * SEQ + n0;
#pragma unroll
        for (int r = 0; r < 8; ++r)
          dst[r] = f2bf(acc[mi][ni][r]);
      }
    }
  }
}

// ---------------------------------------------------------------- proj GEMM
__global__ __launch_bounds__(256)
void proj_gemm(const uint16_t* __restrict__ X, const uint16_t* __restrict__ Wt,
               float* __restrict__ out) {
  __shared__ __align__(16) uint16_t As[2][128 * LDSA];
  __shared__ __align__(16) uint16_t Bt[2][128 * LDSA];

  const int tid  = threadIdx.x;
  const int wave = tid >> 5, lane = tid & 31;
  const int lrow = lane & 15, khalf = lane >> 4;
  const int wm = wave & 3, wn = wave >> 2;
  const int rowBase = blockIdx.y * 128;
  const int colBase = blockIdx.x * 128;

  v8f acc[2][4] = {};
  gemm_mainloop(X, Wt, As[0], Bt[0], rowBase, colBase, acc);

#pragma unroll
  for (int mi = 0; mi < 2; ++mi) {
    int row0 = rowBase + wm * 32 + mi * 16 + khalf * 8;
#pragma unroll
    for (int ni = 0; ni < 4; ++ni) {
      int col = colBase + wn * 64 + ni * 16 + lrow;
      float* dst = out + (size_t)row0 * D_MODEL + col;
#pragma unroll
      for (int r = 0; r < 8; ++r)
        dst[(size_t)r * D_MODEL] = acc[mi][ni][r];
    }
  }
}

// ---------------------------------------------------------------- flash attention
__global__ __launch_bounds__(256)
void flash_attn(const uint16_t* __restrict__ Q, const uint16_t* __restrict__ K,
                const uint16_t* __restrict__ VT, const float* __restrict__ bias,
                const float* __restrict__ beta, uint16_t* __restrict__ O) {
  __shared__ __align__(16) uint16_t pbuf[2][8][16 * 34];  // 34: conflict-free stride

  const int tid  = threadIdx.x;
  const int wave = tid >> 5, lane = tid & 31;
  const int lrow = lane & 15, khalf = lane >> 4;
  const int b = blockIdx.z, h = blockIdx.y;
  const int qbase = blockIdx.x * 128 + wave * 16;

  const uint16_t* qp = Q  + (size_t)(b * N_HEADS + h) * SEQ * D_HEAD;
  const uint16_t* kp = K  + (size_t)(b * N_HEADS + h) * SEQ * D_HEAD;
  const uint16_t* vp = VT + (size_t)(b * N_HEADS + h) * D_HEAD * SEQ;
  const float*    bp = bias + (size_t)b * SEQ * SEQ;
  const bool  has_bias = (h < N_PHYS);
  const float betav    = has_bias ? beta[h] : 0.0f;

  AFrag qf[2];
#pragma unroll
  for (int half = 0; half < 2; ++half)
#pragma unroll
    for (int i = 0; i < 8; ++i) {
      int k = half * 32 + ((i >> 2) << 4) + (khalf << 3) + ((i & 3) << 1);
      qf[half].u[i] = *reinterpret_cast<const uint32_t*>(
          qp + (size_t)(qbase + lrow) * D_HEAD + k);
    }

  float mrun[8], lrun[8];
#pragma unroll
  for (int r = 0; r < 8; ++r) { mrun[r] = -3.0e38f; lrun[r] = 0.0f; }
  v8f acc[4] = {};

  for (int kt = 0; kt < SEQ; kt += 32) {
    if (kt + 32 < SEQ) {
      __builtin_prefetch(kp + (size_t)(kt + 32 + lrow) * D_HEAD, 0, 0);
      __builtin_prefetch(vp + (size_t)lrow * SEQ + kt + 32, 0, 0);
    }
    // ---- S = Q @ K^T for 32 keys
    v8f st[2];
#pragma unroll
    for (int t = 0; t < 2; ++t) {
      v8f s = {};
#pragma unroll
      for (int half = 0; half < 2; ++half) {
        AFrag bf_;
#pragma unroll
        for (int j = 0; j < 8; ++j) {
          int kk = ((j >> 2) << 4) + (khalf << 3) + ((j & 3) << 1);
          bf_.u[j] = *reinterpret_cast<const uint32_t*>(
              kp + (size_t)(kt + t * 16 + lrow) * D_HEAD + half * 32 + kk);
        }
        s = __builtin_amdgcn_wmma_f32_16x16x32_bf16(
            false, qf[half].v, false, bf_.v, (short)0, s, false, false);
      }
      st[t] = s;
    }
    // ---- additive bias on physical heads
    if (has_bias) {
#pragma unroll
      for (int t = 0; t < 2; ++t) {
        const float* bb = bp + (size_t)(qbase + khalf * 8) * SEQ + kt + t * 16 + lrow;
#pragma unroll
        for (int r = 0; r < 8; ++r)
          st[t][r] += betav * bb[(size_t)r * SEQ];
      }
    }
    // ---- online softmax
    float pm[8];
#pragma unroll
    for (int r = 0; r < 8; ++r) pm[r] = fmaxf(st[0][r], st[1][r]);
#pragma unroll
    for (int off = 1; off < 16; off <<= 1)
#pragma unroll
      for (int r = 0; r < 8; ++r) pm[r] = fmaxf(pm[r], __shfl_xor(pm[r], off, 32));
    float cf[8], rs[8];
#pragma unroll
    for (int r = 0; r < 8; ++r) {
      float nm = fmaxf(mrun[r], pm[r]);
      cf[r] = __expf(mrun[r] - nm);
      mrun[r] = nm;
      float p0 = __expf(st[0][r] - nm);
      float p1 = __expf(st[1][r] - nm);
      st[0][r] = p0; st[1][r] = p1;
      rs[r] = p0 + p1;
    }
#pragma unroll
    for (int off = 1; off < 16; off <<= 1)
#pragma unroll
      for (int r = 0; r < 8; ++r) rs[r] += __shfl_xor(rs[r], off, 32);
#pragma unroll
    for (int r = 0; r < 8; ++r) lrun[r] = lrun[r] * cf[r] + rs[r];
#pragma unroll
    for (int g = 0; g < 4; ++g)
#pragma unroll
      for (int r = 0; r < 8; ++r) acc[g][r] *= cf[r];

    // ---- P: C-layout -> A-layout via per-wave LDS tile (parity buffer)
    uint16_t* pb = pbuf[(kt >> 5) & 1][wave];
#pragma unroll
    for (int r = 0; r < 8; ++r) {
      int row = khalf * 8 + r;
      pb[row * 34 + lrow]      = f2bf(st[0][r]);
      pb[row * 34 + 16 + lrow] = f2bf(st[1][r]);
    }
    asm volatile("s_wait_dscnt 0" ::: "memory");
    AFrag pf;
#pragma unroll
    for (int i = 0; i < 8; ++i) {
      int k = ((i >> 2) << 4) + (khalf << 3) + ((i & 3) << 1);
      pf.u[i] = *reinterpret_cast<const uint32_t*>(pb + lrow * 34 + k);
    }
    asm volatile("" ::: "memory");

    // ---- O += P @ V
#pragma unroll
    for (int g = 0; g < 4; ++g) {
      AFrag vf;
#pragma unroll
      for (int j = 0; j < 8; ++j) {
        int kk = ((j >> 2) << 4) + (khalf << 3) + ((j & 3) << 1);
        vf.u[j] = *reinterpret_cast<const uint32_t*>(
            vp + (size_t)(g * 16 + lrow) * SEQ + kt + kk);
      }
      acc[g] = __builtin_amdgcn_wmma_f32_16x16x32_bf16(
          false, pf.v, false, vf.v, (short)0, acc[g], false, false);
    }
  }

  float inv[8];
#pragma unroll
  for (int r = 0; r < 8; ++r) inv[r] = 1.0f / lrun[r];
#pragma unroll
  for (int g = 0; g < 4; ++g) {
    uint16_t* dst = O + (size_t)(b * SEQ + qbase + khalf * 8) * D_MODEL +
                    h * D_HEAD + g * 16 + lrow;
#pragma unroll
    for (int r = 0; r < 8; ++r)
      dst[(size_t)r * D_MODEL] = f2bf(acc[g][r] * inv[r]);
  }
}

// ---------------------------------------------------------------- launch
extern "C" void kernel_launch(void* const* d_in, const int* in_sizes, int n_in,
                              void* d_out, int out_size, void* d_ws, size_t ws_size,
                              hipStream_t stream) {
  (void)in_sizes; (void)n_in; (void)out_size; (void)ws_size;
  const float* x     = (const float*)d_in[0];  // [2,2048,1024]
  const float* bias  = (const float*)d_in[1];  // [2,1,2048,2048]
  const float* wqkv  = (const float*)d_in[2];  // [1024,3072]
  const float* wproj = (const float*)d_in[3];  // [1024,1024]
  const float* beta  = (const float*)d_in[4];  // [8]
  float* out = (float*)d_out;

  uint16_t* xh     = (uint16_t*)d_ws;                          // [4096,1024]
  uint16_t* wqkvt  = xh    + (size_t)ROWS * D_MODEL;           // [3072,1024] = Wqkv^T
  uint16_t* wprojt = wqkvt + (size_t)3 * D_MODEL * D_MODEL;    // [1024,1024] = Wproj^T
  uint16_t* qh     = wprojt + (size_t)D_MODEL * D_MODEL;
  uint16_t* kh     = qh + (size_t)BATCH * N_HEADS * SEQ * D_HEAD;
  uint16_t* vT     = kh + (size_t)BATCH * N_HEADS * SEQ * D_HEAD;
  uint16_t* ah     = vT + (size_t)BATCH * N_HEADS * SEQ * D_HEAD;

  cvt_f32_bf16<<<2048, 256, 0, stream>>>(x, xh, ROWS * D_MODEL);
  transpose_cvt<<<dim3(3 * D_MODEL / 32, D_MODEL / 32), 256, 0, stream>>>(
      wqkv, wqkvt, D_MODEL, 3 * D_MODEL);
  transpose_cvt<<<dim3(D_MODEL / 32, D_MODEL / 32), 256, 0, stream>>>(
      wproj, wprojt, D_MODEL, D_MODEL);

  qkv_gemm<<<dim3(3 * D_MODEL / 128, ROWS / 128), 256, 0, stream>>>(
      xh, wqkvt, qh, kh, vT);

  flash_attn<<<dim3(SEQ / 128, N_HEADS, BATCH), 256, 0, stream>>>(
      qh, kh, vT, bias, beta, ah);

  proj_gemm<<<dim3(D_MODEL / 128, ROWS / 128), 256, 0, stream>>>(
      ah, wprojt, out);
}
